// PILEvolutionTracker_39170101740210
// MI455X (gfx1250) — compile-verified, matrix operand order checked by reading, not run on hardware
//
#include <hip/hip_runtime.h>
#include <hip/hip_bf16.h>
#include <math.h>

// ---------------------------------------------------------------------------
// PILEvolutionTracker for MI455X (gfx1250, wave32)
//
// The reference sorts 2x 1M floats per frame; we replace the sorts with
// histogram selection (monotone float-bit bucketing, 2048 bins) so the whole
// thing runs at the 2-streaming-pass HBM floor (~2 x 256MB / 23.3TB/s).
// Both passes stage frame rows into LDS with gfx1250 async global->LDS
// (global_load_async_to_lds_b128) in a 3-row software pipeline fenced by
// s_wait_asynccnt<=1 (async loads complete in order, so waiting until only
// the newest transfer is outstanding guarantees rows y and y+1 are resident).
// Pass 1: wave-parity-replicated LDS histograms (ds_add_u32).
// Pass 2: masked reductions (count, sum w, sum w*x, sum w*y), w = gm*mask.
// Epilogue: depthwise conv + WMMA (v_wmma_f32_16x16x32_f16) trend dots.
// ---------------------------------------------------------------------------

#define T_FRAMES 64
#define H_DIM 1024
#define W_DIM 1024
#define NPIX (H_DIM * W_DIM)
#define NB 2048
// rank (1-indexed) targets: int(0.85*(N-1))+1 and int(0.30*(N-1))+1
#define TGT_GM 891289u
#define TGT_BZ 314573u

typedef __attribute__((ext_vector_type(16))) _Float16 v16h;
typedef __attribute__((ext_vector_type(8)))  float    v8f;

// Pointee type must match the async-LDS builtin prototype exactly:
// 'int __attribute__((vector_size(4*sizeof(int))))' in AS1 / AS3.
typedef int v4i_ __attribute__((vector_size(4 * sizeof(int))));
typedef __attribute__((address_space(1))) v4i_ gas_v4i;
typedef __attribute__((address_space(3))) v4i_ las_v4i;

#if __has_builtin(__builtin_amdgcn_global_load_async_to_lds_b128)
#define HAVE_ASYNC_LDS 1
#else
#define HAVE_ASYNC_LDS 0
#endif

// Monotone bucket for v >= 0: float bits >> 20 = {sign=0, exp[8], mant[3]}.
__device__ __forceinline__ unsigned bucket_of(float v) {
  unsigned b = __float_as_uint(v) >> 20;
  return b < (NB - 1u) ? b : (NB - 1u);
}

// One 128-bit async transfer per thread covers a 1024-float row with 256 thr.
__device__ __forceinline__ void stage_row(const float* __restrict__ src,
                                          float* dst, int tid, bool valid) {
  if (!valid) return;                         // block-uniform
#if HAVE_ASYNC_LDS
  for (int i = tid * 4; i < W_DIM; i += 256 * 4) {
    __builtin_amdgcn_global_load_async_to_lds_b128(
        (gas_v4i*)(src + i), (las_v4i*)(dst + i), 0, 0);
  }
#else
  for (int i = tid * 4; i < W_DIM; i += 256 * 4)
    *(float4*)(dst + i) = *(const float4*)(src + i);
#endif
}

// Wait until at most n async transfers from this wave are outstanding.
__device__ __forceinline__ void async_fence_n(int n) {
#if HAVE_ASYNC_LDS
#if __has_builtin(__builtin_amdgcn_s_wait_asynccnt)
  if (n == 0) __builtin_amdgcn_s_wait_asynccnt(0);
  else        __builtin_amdgcn_s_wait_asynccnt(1);
#else
  if (n == 0) asm volatile("s_wait_asynccnt 0x0" ::: "memory");
  else        asm volatile("s_wait_asynccnt 0x1" ::: "memory");
#endif
#endif
}

// ---------------------------------------------------------------------------
__global__ void k_zero(unsigned* __restrict__ ws, int nwords) {
  int i = blockIdx.x * 256 + threadIdx.x;
  if (i < nwords) ws[i] = 0u;
}

// ---------------------------------------------------------------------------
// Pass 1: per-frame histograms of grad-magnitude and |f|.
// grid = (32 slabs, 64 frames), 256 threads. Async-staged rows, 3-deep
// pipeline; 2x wave-parity-replicated LDS histograms to cut ds_add conflicts.
__global__ void k_hist(const float* __restrict__ F, unsigned* __restrict__ histG,
                       unsigned* __restrict__ histB) {
  __shared__ float rowbuf[3][W_DIM];          // 12 KB
  __shared__ unsigned hG[2][NB];              // 16 KB
  __shared__ unsigned hB[2][NB];              // 16 KB
  const int tid = threadIdx.x;
  for (int i = tid; i < 2 * NB; i += 256) {
    ((unsigned*)hG)[i] = 0u;
    ((unsigned*)hB)[i] = 0u;
  }
  const int t    = blockIdx.y;
  const int slab = blockIdx.x * 32;
  const size_t base = (size_t)t * NPIX;
  const int rep = (tid >> 5) & 1;             // histogram replica by wave parity

  stage_row(F + base + (size_t)(slab + 0) * W_DIM, rowbuf[0], tid, true);
  stage_row(F + base + (size_t)(slab + 1) * W_DIM, rowbuf[1], tid, true);

  for (int r = 0; r < 32; ++r) {
    const int y = slab + r;
    const bool ahead = (r < 31) && (y + 2 < H_DIM);
    stage_row(F + base + (size_t)(y + 2) * W_DIM, rowbuf[(r + 2) % 3], tid, ahead);
    async_fence_n(ahead ? 1 : 0);             // rows y, y+1 now resident
    __syncthreads();                          // also covers histogram zeroing
    const float* rc = rowbuf[r % 3];
    const float* rn = rowbuf[(r + 1) % 3];
    const bool haveNext = (y + 1) < H_DIM;
    for (int x = tid; x < W_DIM; x += 256) {
      const float v  = rc[x];
      const float gx = (x < W_DIM - 1) ? rc[x + 1] - v : 0.f;  // zero-pad right
      const float gy = haveNext ? rn[x] - v : 0.f;             // zero-pad bottom
      const float gm = sqrtf(gx * gx + gy * gy);
      atomicAdd(&hG[rep][bucket_of(gm)], 1u);                  // ds_add_u32
      atomicAdd(&hB[rep][bucket_of(fabsf(v))], 1u);
    }
    __syncthreads();  // protect rowbuf before next iteration overwrites it
  }

  unsigned* gG = histG + (size_t)t * NB;
  unsigned* gB = histB + (size_t)t * NB;
  for (int i = tid; i < NB; i += 256) {
    const unsigned cg = hG[0][i] + hG[1][i];
    const unsigned cb = hB[0][i] + hB[1][i];
    if (cg) atomicAdd(&gG[i], cg);
    if (cb) atomicAdd(&gB[i], cb);
  }
}

// ---------------------------------------------------------------------------
// Per-frame quantile thresholds: serial scan over 2048 bins (tiny).
// thr[t] = gm threshold, thr[64+t] = |f| threshold (bucket lower edges).
__global__ void k_thresh(const unsigned* __restrict__ histG,
                         const unsigned* __restrict__ histB,
                         float* __restrict__ thr) {
  const int t = threadIdx.x;
  if (t >= T_FRAMES) return;
  const unsigned* hG = histG + (size_t)t * NB;
  const unsigned* hB = histB + (size_t)t * NB;
  unsigned cg = 0, cb = 0;
  int bg = NB - 1, bb = NB - 1;
  bool fg = false, fb = false;
  for (int i = 0; i < NB; ++i) {
    __builtin_prefetch(hG + i + 32, 0, 1);    // global_prefetch_b8
    cg += hG[i];
    if (!fg && cg >= TGT_GM) { bg = i; fg = true; }
    cb += hB[i];
    if (!fb && cb >= TGT_BZ) { bb = i; fb = true; }
  }
  thr[t]            = __uint_as_float((unsigned)bg << 20);
  thr[T_FRAMES + t] = __uint_as_float((unsigned)bb << 20);
}

// ---------------------------------------------------------------------------
// Pass 2: masked reductions per frame: {count, sum(w), sum(w*x), sum(w*y)},
// w = gm * mask. Same async-staged 3-row pipeline as pass 1.
__global__ void k_stats(const float* __restrict__ F, const float* __restrict__ thr,
                        float* __restrict__ sums) {
  __shared__ float rowbuf[3][W_DIM];          // 12 KB
  __shared__ float redbuf[8][4];
  const int tid  = threadIdx.x;
  const int t    = blockIdx.y;
  const int slab = blockIdx.x * 32;
  const size_t base = (size_t)t * NPIX;
  const float thrG = thr[t];
  const float thrB = thr[T_FRAMES + t];
  float cnt = 0.f, gsum = 0.f, wx = 0.f, wy = 0.f;

  stage_row(F + base + (size_t)(slab + 0) * W_DIM, rowbuf[0], tid, true);
  stage_row(F + base + (size_t)(slab + 1) * W_DIM, rowbuf[1], tid, true);

  for (int r = 0; r < 32; ++r) {
    const int y = slab + r;
    const bool ahead = (r < 31) && (y + 2 < H_DIM);
    stage_row(F + base + (size_t)(y + 2) * W_DIM, rowbuf[(r + 2) % 3], tid, ahead);
    async_fence_n(ahead ? 1 : 0);
    __syncthreads();
    const float* rc = rowbuf[r % 3];
    const float* rn = rowbuf[(r + 1) % 3];
    const bool haveNext = (y + 1) < H_DIM;
    const float fy = (float)y;
    for (int x = tid; x < W_DIM; x += 256) {
      const float v  = rc[x];
      const float gx = (x < W_DIM - 1) ? rc[x + 1] - v : 0.f;
      const float gy = haveNext ? rn[x] - v : 0.f;
      const float gm = sqrtf(gx * gx + gy * gy);
      if ((gm > thrG) && (fabsf(v) < thrB)) {
        cnt += 1.f; gsum += gm; wx += gm * (float)x; wy += gm * fy;
      }
    }
    __syncthreads();
  }

  // wave32 cross-lane reduction, then cross-wave via LDS, then global atomics.
  float a0 = cnt, a1 = gsum, a2 = wx, a3 = wy;
  for (int off = 16; off > 0; off >>= 1) {
    a0 += __shfl_down(a0, off, 32);
    a1 += __shfl_down(a1, off, 32);
    a2 += __shfl_down(a2, off, 32);
    a3 += __shfl_down(a3, off, 32);
  }
  const int lane = tid & 31, wv = tid >> 5;
  if (lane == 0) { redbuf[wv][0]=a0; redbuf[wv][1]=a1; redbuf[wv][2]=a2; redbuf[wv][3]=a3; }
  __syncthreads();
  if (tid < 4) {
    float s = 0.f;
    for (int w = 0; w < 8; ++w) s += redbuf[w][tid];
    atomicAdd(&sums[(size_t)t * 4 + tid], s);
  }
}

// ---------------------------------------------------------------------------
// Epilogue: per-frame stats, depthwise conv (k=3, pad=1, cross-correlation),
// WMMA trend dot-products, final 8 outputs. 1 block, 64 threads (2 waves).
__global__ void k_final(const float* __restrict__ sums, const float* __restrict__ cw,
                        const float* __restrict__ cb, float* __restrict__ out) {
  __shared__ float raw[4][66];                // +1 zero pad each side
  __shared__ float sX[64], sY[64];
  __shared__ float Atab[16][64];              // zero-padded A rows (branch-free)
  __shared__ float Btab[16][64];              // zero-padded B cols (branch-free)
  __shared__ float res[2];
  const int t = threadIdx.x;                  // 0..63

  for (int i = t; i < 16 * 64; i += 64) {
    ((float*)Atab)[i] = 0.f;
    ((float*)Btab)[i] = 0.f;
  }

  const float c  = sums[t * 4 + 0];
  const float g  = sums[t * 4 + 1];
  const float wx = sums[t * 4 + 2];
  const float wy = sums[t * 4 + 3];
  const float pil  = fmaxf(c, 1e-6f);
  const float wsum = fmaxf(g, 1e-6f);
  raw[0][t + 1] = pil / (float)NPIX;          // length
  raw[1][t + 1] = g / pil;                    // intensity
  raw[2][t + 1] = wx / wsum / (float)W_DIM;   // x centroid
  raw[3][t + 1] = wy / wsum / (float)H_DIM;   // y centroid
  if (t == 0)  { raw[0][0]  = raw[1][0]  = raw[2][0]  = raw[3][0]  = 0.f; }
  if (t == 63) { raw[0][65] = raw[1][65] = raw[2][65] = raw[3][65] = 0.f; }
  __syncthreads();

  // y[c,t] = sum_k w[c,0,k] * x[c, t+k-1] + b[c]  (JAX conv = cross-correlation)
  Atab[0][t] = cw[0]*raw[0][t] + cw[1] *raw[0][t+1] + cw[2] *raw[0][t+2] + cb[0]; // length
  Atab[1][t] = cw[3]*raw[1][t] + cw[4] *raw[1][t+1] + cw[5] *raw[1][t+2] + cb[1]; // intensity
  sX[t]      = cw[6]*raw[2][t] + cw[7] *raw[2][t+1] + cw[8] *raw[2][t+2] + cb[2];
  sY[t]      = cw[9]*raw[3][t] + cw[10]*raw[3][t+1] + cw[11]*raw[3][t+2] + cb[3];
  Btab[1][t] = (float)t * (1.f / 63.f) - 0.5f;  // trend weights in column 1
  __syncthreads();

  // Trend dot-products on the matrix core (wave 0 only -> EXEC all 1s):
  // A row0 = length series, row1 = intensity series (two K=32 chunks),
  // B col1 = (t/63 - 0.5). Sum of weights is exactly 0, so trend = D[m,1]*6/n.
  // f16 inputs on O(1) stats, f32 accumulation.
  if (t < 32) {
    const int L = t, m = L & 15;
    v8f acc = {};
    for (int chunk = 0; chunk < 2; ++chunk) {
      v16h a, b;
      for (int h = 0; h < 16; ++h) {
        const int vv = h >> 1, pos = h & 1;
        // 16-bit A layout: lane<16 holds K 0-7 (v0-3) & 16-23 (v4-7); +8 for lane>=16
        const int k  = ((vv & 3) * 2 + pos) + (L >= 16 ? 8 : 0) + (vv >= 4 ? 16 : 0);
        const int tt = chunk * 32 + k;
        a[h] = (_Float16)Atab[m][tt];
        b[h] = (_Float16)Btab[m][tt];
      }
      acc = __builtin_amdgcn_wmma_f32_16x16x32_f16(false, a, false, b,
                                                   (short)0, acc, false, false);
    }
    // D layout: VGPR j, lanes 0-15 -> M=j, N=lane. Need D[0,1], D[1,1].
    if (L == 1) { res[0] = acc[0]; res[1] = acc[1]; }
  }
  __syncthreads();

  if (t == 0) {
    float lm = 0.f, im = 0.f;
    for (int i = 0; i < 64; ++i) { lm += Atab[0][i]; im += Atab[1][i]; }
    lm *= (1.f / 64.f); im *= (1.f / 64.f);
    const float ltr = res[0] * 6.f / 64.f;
    const float itr = res[1] * 6.f / 64.f;
    float dxs = 0.f, dys = 0.f, sps = 0.f, sp2 = 0.f;
    for (int i = 0; i < 63; ++i) {
      const float dx = sX[i + 1] - sX[i];
      const float dy = sY[i + 1] - sY[i];
      const float sp = sqrtf(dx * dx + dy * dy);
      dxs += dx; dys += dy; sps += sp; sp2 += sp * sp;
    }
    const float ms  = sps / 63.f;
    const float dir = atan2f(dys, dxs) / 3.14159265358979323846f;
    const float gr  = (Atab[0][63] - Atab[0][0]) / 63.f;
    const float var = fmaxf((sp2 - 63.f * ms * ms) / 62.f, 0.f);  // ddof=1
    out[0] = lm;  out[1] = ltr; out[2] = im;  out[3] = itr;
    out[4] = ms;  out[5] = dir; out[6] = gr;  out[7] = sqrtf(var);
  }
}

// ---------------------------------------------------------------------------
extern "C" void kernel_launch(void* const* d_in, const int* in_sizes, int n_in,
                              void* d_out, int out_size, void* d_ws, size_t ws_size,
                              hipStream_t stream) {
  const float* frames = (const float*)d_in[0];
  // d_in[1] = observed_mask: unused by the reference math.
  const float* conv_w = (const float*)d_in[2];  // (4,1,3) row-major
  const float* conv_b = (const float*)d_in[3];  // (4,)
  float* out = (float*)d_out;

  unsigned* histG = (unsigned*)d_ws;
  unsigned* histB = histG + (size_t)T_FRAMES * NB;
  float*    thr   = (float*)(histB + (size_t)T_FRAMES * NB);
  float*    sums  = thr + 2 * T_FRAMES;

  const int nwords = T_FRAMES * NB * 2 + 2 * T_FRAMES + 4 * T_FRAMES;
  k_zero<<<(nwords + 255) / 256, 256, 0, stream>>>((unsigned*)d_ws, nwords);
  k_hist<<<dim3(32, T_FRAMES), 256, 0, stream>>>(frames, histG, histB);
  k_thresh<<<1, 64, 0, stream>>>(histG, histB, thr);
  k_stats<<<dim3(32, T_FRAMES), 256, 0, stream>>>(frames, thr, sums);
  k_final<<<1, 64, 0, stream>>>(sums, conv_w, conv_b, out);
}
